// MultiHeadAttention_747324309937
// MI455X (gfx1250) — compile-verified, matrix operand order checked by reading, not run on hardware
//
#include <hip/hip_runtime.h>
#include <math.h>

#define DIMSZ 1024
#define NHEAD 16
#define DK    64
#define SEQT  2048
#define BATCH 4

// ---------------- WMMA types / helpers (CDNA5 gfx1250, wave32) ----------------
typedef __attribute__((ext_vector_type(16))) __bf16 v16bf;
typedef __attribute__((ext_vector_type(2)))  __bf16 v2bf;
typedef __attribute__((ext_vector_type(8)))  float  v8f;
typedef __attribute__((ext_vector_type(4)))  unsigned v4u;
typedef __attribute__((ext_vector_type(8)))  int   v8i;
typedef __attribute__((ext_vector_type(4)))  int   v4i;

union BFVec { v16bf v; unsigned short u[16]; };

__device__ __forceinline__ v8f vzero8() {
  v8f z = {0.f,0.f,0.f,0.f,0.f,0.f,0.f,0.f};
  return z;
}

__device__ __forceinline__ unsigned short f2bf(float f) {
  unsigned u = __builtin_bit_cast(unsigned, f);
  unsigned r = u + 0x7FFFu + ((u >> 16) & 1u);   // round-to-nearest-even
  return (unsigned short)(r >> 16);
}

// packed f32x2 -> bf16x2 (one v_cvt_pk_bf16_f32 when available)
__device__ __forceinline__ unsigned f2bf_pk(float lo, float hi) {
#if __has_builtin(__builtin_amdgcn_cvt_pk_bf16_f32)
  v2bf p = __builtin_amdgcn_cvt_pk_bf16_f32(lo, hi);
  return __builtin_bit_cast(unsigned, p);
#else
  return (unsigned)f2bf(lo) | ((unsigned)f2bf(hi) << 16);
#endif
}

// ---- gfx1250 async global->LDS copy (ASYNCcnt path), via inline asm ----
__device__ __forceinline__ void async_b128(unsigned lds_addr, const void* gaddr) {
  asm volatile("global_load_async_to_lds_b128 %0, %1, off"
               :: "v"(lds_addr), "v"(gaddr) : "memory");
}
__device__ __forceinline__ void wait_async0() {
  asm volatile("s_wait_asynccnt 0x0" ::: "memory");
}
__device__ __forceinline__ void async_copy64B(unsigned short* ldsDst, const unsigned short* gSrc) {
#pragma unroll
  for (int i = 0; i < 4; ++i)
    async_b128((unsigned)(uintptr_t)(ldsDst + i*8), (const void*)(gSrc + i*8));
}

// ---- gfx1250 Tensor Data Mover: one-instruction 64x64 bf16 tile load ----
__device__ __forceinline__ void wait_tensor0() {
#if __has_builtin(__builtin_amdgcn_s_wait_tensorcnt)
  __builtin_amdgcn_s_wait_tensorcnt(0);
#else
  asm volatile("s_wait_tensorcnt 0x0" ::: "memory");
#endif
}

#if __has_builtin(__builtin_amdgcn_tensor_load_to_lds)
#define HAVE_TDM 1
// D# for a 2D tile: 64x64 elements, data_size=2B, row stride 64 elements,
// tensor dims == tile dims (no OOB), dense LDS destination (stride 64).
__device__ __forceinline__ void tdm_load_tile_64x64_bf16(unsigned lds_addr, const void* gaddr) {
  const unsigned long long ga = (unsigned long long)(uintptr_t)gaddr;
  // group0: [1:0]=count=1 | [63:32]=lds_addr | [120:64]=global_addr | [127:126]=type=2
  v4u g0 = { 1u, lds_addr,
             (unsigned)ga,
             (unsigned)((ga >> 32) & 0x01FFFFFFu) | 0x80000000u };
  // group1: data_size=1(2B); tensor_dim0=64 @bit48; tensor_dim1=64 @bit80;
  //         tile_dim0=64 @bit112; tile_dim1=64 @bit128; dim0_stride=64 @bit160
  v8i g1 = { 0x10000, 0x400000, 0x400000, 0x400000, 64, 64, 0, 0 };
  v4i gz = { 0, 0, 0, 0 };
#if __has_include(<hip/amd_detail/amd_gfx1250_TDM.h>)
  v8i gz8 = { 0,0,0,0,0,0,0,0 };
  __builtin_amdgcn_tensor_load_to_lds(g0, g1, gz, gz, gz8, 0);   // clang-23 6-arg form
#else
  __builtin_amdgcn_tensor_load_to_lds(g0, g1, gz, gz, 0);        // ROCm 7.2 5-arg form
#endif
}
#else
#define HAVE_TDM 0
#endif

// Fragment element (m,k) = tile[(m0+m)*ld + k0+k]  (16x32 bf16, row-major along K)
// ISA 16-bit A-matrix layout: lanes 0-15 hold K=0-7 & 16-23, lanes 16-31 K=8-15 & 24-31.
__device__ __forceinline__ v16bf load_frag_rowK(const unsigned short* t, int ld, int m0, int k0) {
  const int lane = threadIdx.x & 31;
  const int m  = m0 + (lane & 15);
  const int kb = (lane >> 4) << 3;               // 0 or 8
  BFVec f;
#pragma unroll
  for (int i = 0; i < 8; ++i) f.u[i]     = t[m*ld + k0 + kb + i];
#pragma unroll
  for (int i = 0; i < 8; ++i) f.u[8 + i] = t[m*ld + k0 + 16 + kb + i];
  return f.v;
}

// Fragment element (k,n) = tile[(k0+k)*ld + n0+n]  (B operand stored K-major)
__device__ __forceinline__ v16bf load_frag_colN(const unsigned short* t, int ld, int n0, int k0) {
  const int lane = threadIdx.x & 31;
  const int n  = n0 + (lane & 15);
  const int kb = (lane >> 4) << 3;
  BFVec f;
#pragma unroll
  for (int i = 0; i < 8; ++i) f.u[i]     = t[(k0 + kb + i)*ld + n];
#pragma unroll
  for (int i = 0; i < 8; ++i) f.u[8 + i] = t[(k0 + 16 + kb + i)*ld + n];
  return f.v;
}

// ---------------- GEMM:  C[M,N] = A[M,K] * W[N,K]^T + bias ----------------
#define BM  64
#define BN  128
#define BK  32
#define LDT 34   // padded LDS stride (ushorts, 4B-aligned rows)

__global__ __launch_bounds__(256)
void gemm_xwT(const float* __restrict__ A32, const unsigned short* __restrict__ A16,
              const float* __restrict__ W, const float* __restrict__ bias,
              unsigned short* __restrict__ ChBF, float* __restrict__ Cf,
              int M, int K) {
  __shared__ __align__(16) unsigned short As[BM * LDT];
  __shared__ __align__(16) unsigned short Bs[BN * LDT];

  const int tid = threadIdx.x;
  const int bm = blockIdx.y, bn = blockIdx.x;
  const int w  = tid >> 5, wm = w >> 2, wn = w & 3;

  v8f acc[2][2];
#pragma unroll
  for (int i = 0; i < 2; ++i)
#pragma unroll
    for (int j = 0; j < 2; ++j) acc[i][j] = vzero8();

  for (int kt = 0; kt < K; kt += BK) {
    {
      const int r = tid >> 2, c0 = (tid & 3) * 8;
      const size_t g = (size_t)(bm*BM + r) * K + kt + c0;
      unsigned* dst = (unsigned*)&As[r*LDT + c0];
      if (A16) {
        const unsigned* src = (const unsigned*)&A16[g];
#pragma unroll
        for (int i = 0; i < 4; ++i) dst[i] = src[i];
      } else {
#pragma unroll
        for (int i = 0; i < 4; ++i) dst[i] = f2bf_pk(A32[g + 2*i], A32[g + 2*i + 1]);
      }
    }
    {
      const int r = tid >> 1, c0 = (tid & 1) * 16;
      const size_t g = (size_t)(bn*BN + r) * K + kt + c0;
      if (kt + BK < K) __builtin_prefetch(W + g + BK, 0, 1);  // global_prefetch_b8
      unsigned* dst = (unsigned*)&Bs[r*LDT + c0];
#pragma unroll
      for (int i = 0; i < 8; ++i) dst[i] = f2bf_pk(W[g + 2*i], W[g + 2*i + 1]);
    }
    __syncthreads();

    const v16bf a0 = load_frag_rowK(As, LDT, wm*32,      0);
    const v16bf a1 = load_frag_rowK(As, LDT, wm*32 + 16, 0);
    const v16bf b0 = load_frag_rowK(Bs, LDT, wn*32,      0);
    const v16bf b1 = load_frag_rowK(Bs, LDT, wn*32 + 16, 0);
    acc[0][0] = __builtin_amdgcn_wmma_f32_16x16x32_bf16(false, a0, false, b0, (short)0, acc[0][0], false, false);
    acc[0][1] = __builtin_amdgcn_wmma_f32_16x16x32_bf16(false, a0, false, b1, (short)0, acc[0][1], false, false);
    acc[1][0] = __builtin_amdgcn_wmma_f32_16x16x32_bf16(false, a1, false, b0, (short)0, acc[1][0], false, false);
    acc[1][1] = __builtin_amdgcn_wmma_f32_16x16x32_bf16(false, a1, false, b1, (short)0, acc[1][1], false, false);
    __syncthreads();
  }

  const int lane = tid & 31;
  const int halfsel = (lane >> 4) << 3;
  const int cn = lane & 15;
#pragma unroll
  for (int mt = 0; mt < 2; ++mt)
#pragma unroll
    for (int nt = 0; nt < 2; ++nt)
#pragma unroll
      for (int r = 0; r < 8; ++r) {
        const int row = bm*BM + wm*32 + mt*16 + r + halfsel;
        const int col = bn*BN + wn*32 + nt*16 + cn;
        const float v = acc[mt][nt][r] + bias[col];
        if (Cf) {
          Cf[(size_t)row * DIMSZ + col] = v;
        } else {
          const int b = row >> 11;           // /SEQT
          const int t = row & (SEQT - 1);
          const int h = col >> 6;            // /DK
          const int d = col & 63;
          ChBF[((size_t)(b*NHEAD + h) * SEQT + t) * DK + d] = f2bf(v);
        }
      }
}

// ---------------- Flash attention ----------------
// Block = 128 threads (4 waves). K/V tiles: single TDM tensor_load_to_lds per tile
// (TENSORcnt); Q tile: per-lane async global->LDS (ASYNCcnt). Both bypass VGPRs.
#define QT  64
#define QLD 72   // Q/P stride: 144B rows, 16B aligned
#define KLD 64   // K/V stride: dense 128B rows (TDM writes dense tiles)

__global__ __launch_bounds__(128)
void flash_attn(const unsigned short* __restrict__ Qh,
                const unsigned short* __restrict__ Kh,
                const unsigned short* __restrict__ Vh,
                const int* __restrict__ mask,
                unsigned short* __restrict__ AO) {
  __shared__ __align__(16) unsigned short Qs[QT * QLD];
  __shared__ __align__(16) unsigned short Ks[QT * KLD];
  __shared__ __align__(16) unsigned short Vs[QT * KLD];
  __shared__ __align__(16) unsigned short Ps[4][16 * QLD];

  const int tid  = threadIdx.x;
  const int w    = tid >> 5;
  const int lane = tid & 31;
  const int halfsel = (lane >> 4) << 3;
  const int cn = lane & 15;

  const int bh = blockIdx.y;
  const int b  = bh >> 4;           // /NHEAD
  const int h  = bh & 15;
  const int q0 = blockIdx.x * QT;
  const size_t headBase = (size_t)bh * SEQT * DK;

  const int r  = tid >> 1;          // row this thread stages (0..63)
  const int c0 = (tid & 1) * 32;    // 32-ushort (64B) half-row

  // async-stage Q tile (ASYNCcnt; waited inside first loop iteration)
  async_copy64B(&Qs[r*QLD + c0], Qh + headBase + (size_t)(q0 + r)*DK + c0);

  v8f o[4];
  float mi[8], li[8];
#pragma unroll
  for (int i = 0; i < 4; ++i) o[i] = vzero8();
#pragma unroll
  for (int rr = 0; rr < 8; ++rr) { mi[rr] = -1e30f; li[rr] = 0.f; }

  for (int j0 = 0; j0 < SEQT; j0 += QT) {
    __syncthreads();   // previous iteration done reading Ks/Vs
#if HAVE_TDM
    if (w == 0) {      // one wave drives the TDM; EXEC-independent DMA
      tdm_load_tile_64x64_bf16((unsigned)(uintptr_t)&Ks[0],
                               (const void*)(Kh + headBase + (size_t)j0 * DK));
      tdm_load_tile_64x64_bf16((unsigned)(uintptr_t)&Vs[0],
                               (const void*)(Vh + headBase + (size_t)j0 * DK));
      wait_tensor0();
    }
#else
    async_copy64B(&Ks[r*KLD + c0], Kh + headBase + (size_t)(j0 + r)*DK + c0);
    async_copy64B(&Vs[r*KLD + c0], Vh + headBase + (size_t)(j0 + r)*DK + c0);
#endif
    wait_async0();     // Q (iter 0) / fallback K,V async transfers landed
    __syncthreads();   // publish tiles to all waves

    // S[q,s] = sum_d Q[q,d] K[s,d]
    v8f sv[4];
#pragma unroll
    for (int nt = 0; nt < 4; ++nt) {
      sv[nt] = vzero8();
#pragma unroll
      for (int kk = 0; kk < DK; kk += 32) {
        const v16bf a  = load_frag_rowK(Qs, QLD, w*16,  kk);
        const v16bf bb = load_frag_rowK(Ks, KLD, nt*16, kk);
        sv[nt] = __builtin_amdgcn_wmma_f32_16x16x32_bf16(false, a, false, bb, (short)0, sv[nt], false, false);
      }
    }

    // scale + mask (mask==0 -> -60000 on scaled score)
#pragma unroll
    for (int nt = 0; nt < 4; ++nt)
#pragma unroll
      for (int rr = 0; rr < 8; ++rr) {
        const int qrow = q0 + w*16 + rr + halfsel;
        const int kcol = j0 + nt*16 + cn;
        const float s  = sv[nt][rr] * 0.125f;   // 1/sqrt(64)
        const int mv = mask[((size_t)b * SEQT + qrow) * SEQT + kcol];
        sv[nt][rr] = (mv == 0) ? -60000.0f : s;
      }

    // online softmax (row = acc vgpr index, +8 for upper half-wave)
    float rsum[8], alpha[8];
#pragma unroll
    for (int rr = 0; rr < 8; ++rr) {
      float mx = sv[0][rr];
#pragma unroll
      for (int nt = 1; nt < 4; ++nt) mx = fmaxf(mx, sv[nt][rr]);
#pragma unroll
      for (int off = 1; off < 16; off <<= 1) mx = fmaxf(mx, __shfl_xor(mx, off, 16));
      const float nm = fmaxf(mi[rr], mx);
      alpha[rr] = __expf(mi[rr] - nm);
      mi[rr] = nm;
    }
#pragma unroll
    for (int rr = 0; rr < 8; ++rr) rsum[rr] = 0.f;
#pragma unroll
    for (int nt = 0; nt < 4; ++nt)
#pragma unroll
      for (int rr = 0; rr < 8; ++rr) {
        const float p = __expf(sv[nt][rr] - mi[rr]);
        sv[nt][rr] = p;
        rsum[rr] += p;
      }
#pragma unroll
    for (int rr = 0; rr < 8; ++rr) {
#pragma unroll
      for (int off = 1; off < 16; off <<= 1) rsum[rr] += __shfl_xor(rsum[rr], off, 16);
      li[rr] = li[rr]*alpha[rr] + rsum[rr];
    }
#pragma unroll
    for (int dt = 0; dt < 4; ++dt)
#pragma unroll
      for (int rr = 0; rr < 8; ++rr) o[dt][rr] *= alpha[rr];

    // P: C-layout regs -> bf16 A-layout via per-wave LDS tile (DS in-order per wave)
#pragma unroll
    for (int nt = 0; nt < 4; ++nt)
#pragma unroll
      for (int rr = 0; rr < 8; ++rr)
        Ps[w][(rr + halfsel)*QLD + nt*16 + cn] = f2bf(sv[nt][rr]);

    // O[q,d] += sum_s P[q,s] V[s,d]
#pragma unroll
    for (int dt = 0; dt < 4; ++dt)
#pragma unroll
      for (int kk = 0; kk < QT; kk += 32) {
        const v16bf a  = load_frag_rowK(&Ps[w][0], QLD, 0, kk);
        const v16bf bb = load_frag_colN(Vs, KLD, dt*16, kk);
        o[dt] = __builtin_amdgcn_wmma_f32_16x16x32_bf16(false, a, false, bb, (short)0, o[dt], false, false);
      }
  }

  // normalize; store attention output [B*T, DIM] bf16 (input to Wo GEMM)
#pragma unroll
  for (int rr = 0; rr < 8; ++rr) li[rr] = 1.0f / li[rr];
#pragma unroll
  for (int dt = 0; dt < 4; ++dt)
#pragma unroll
    for (int rr = 0; rr < 8; ++rr) {
      const int qrow = q0 + w*16 + rr + halfsel;
      const int dcol = h*DK + dt*16 + cn;
      AO[((size_t)b * SEQT + qrow) * DIMSZ + dcol] = f2bf(o[dt][rr] * li[rr]);
    }
}

// ---------------- Host launch ----------------
extern "C" void kernel_launch(void* const* d_in, const int* in_sizes, int n_in,
                              void* d_out, int out_size, void* d_ws, size_t ws_size,
                              hipStream_t stream) {
  const float* q    = (const float*)d_in[0];
  const float* k    = (const float*)d_in[1];
  const float* v    = (const float*)d_in[2];
  const int*   mask = (const int*)  d_in[3];
  const float* Wq   = (const float*)d_in[4];
  const float* bq   = (const float*)d_in[5];
  const float* Wk   = (const float*)d_in[6];
  const float* bk   = (const float*)d_in[7];
  const float* Wv   = (const float*)d_in[8];
  const float* bv   = (const float*)d_in[9];
  const float* Wo   = (const float*)d_in[10];
  const float* bo   = (const float*)d_in[11];
  float* out = (float*)d_out;

  const size_t headElems = (size_t)BATCH * NHEAD * SEQT * DK;   // 8M bf16 each
  unsigned short* Qh = (unsigned short*)d_ws;
  unsigned short* Kh = Qh + headElems;
  unsigned short* Vh = Kh + headElems;
  unsigned short* AO = Vh + headElems;                          // [B*T, DIM] bf16

  const int M = BATCH * SEQT;                                   // 8192
  dim3 gg(DIMSZ / BN, M / BM);                                  // (8, 128)
  gemm_xwT<<<gg, 256, 0, stream>>>(q, nullptr, Wq, bq, Qh, nullptr, M, DIMSZ);
  gemm_xwT<<<gg, 256, 0, stream>>>(k, nullptr, Wk, bk, Kh, nullptr, M, DIMSZ);
  gemm_xwT<<<gg, 256, 0, stream>>>(v, nullptr, Wv, bv, Vh, nullptr, M, DIMSZ);

  dim3 ga(SEQT / QT, BATCH * NHEAD);                            // (32, 64)
  flash_attn<<<ga, 128, 0, stream>>>(Qh, Kh, Vh, mask, AO);

  gemm_xwT<<<gg, 256, 0, stream>>>(nullptr, AO, Wo, bo, nullptr, out, M, DIMSZ);
}
